// Sub_Cluster_Level_GCN_63745904607636
// MI455X (gfx1250) — compile-verified, matrix operand order checked by reading, not run on hardware
//
#include <hip/hip_runtime.h>

// ---------------- problem constants ----------------
#define B_    256
#define K_    64
#define N_    50000
#define D_    2048
#define S_    2000
#define NHID_ 512
#define DOUT_ 256

// ---------------- CDNA5 WMMA types ----------------
typedef __bf16 v16bf __attribute__((ext_vector_type(16)));
typedef float  v8f   __attribute__((ext_vector_type(8)));

static __device__ __forceinline__ unsigned short f2b(float f) {
  unsigned int u = __float_as_uint(f);
  u += 0x7FFFu + ((u >> 16) & 1u);           // round-to-nearest-even
  return (unsigned short)(u >> 16);
}
static __device__ __forceinline__ float b2f(unsigned short h) {
  return __uint_as_float(((unsigned int)h) << 16);
}

static __device__ __forceinline__ v8f wmma_bf16(v16bf a, v16bf b, v8f c) {
  // D = A(16x32 bf16) * B(32x16 bf16) + C(16x16 f32)
  return __builtin_amdgcn_wmma_f32_16x16x32_bf16(false, a, false, b,
                                                 (short)0, c, false, false);
}

// Build an A/B fragment from an LDS row (row-major, >=32 bf16 of K starting
// at p).  Per the ISA 16-bit layout, lane (half = lane>>4) holds K =
// half*8+{0..7} and 16+half*8+{0..7}: two b128 loads.
static __device__ __forceinline__ v16bf frag_ld(const unsigned short* p, int half) {
  union { v16bf v; float4 f[2]; } u;
  u.f[0] = *(const float4*)(p + half * 8);
  u.f[1] = *(const float4*)(p + 16 + half * 8);
  return u.v;
}

// ---------------- CDNA5 async global->LDS copy (ASYNCcnt path) ----------------
// GVS addressing: mem = SADDR(64b, uniform) + VADDR(32b, per-lane byte offset);
// LDS dest = LDS_BASE + VDST (per-lane byte offset within workgroup LDS).
static __device__ __forceinline__ unsigned lds_off(const void* p) {
  return (unsigned)(uintptr_t)p;             // LDS offset is the low 32 bits of the address
}
static __device__ __forceinline__ void async_ld_b128(unsigned ldsdst,
                                                     const void* sbase,
                                                     unsigned voff) {
  asm volatile("global_load_async_to_lds_b128 %0, %1, %2"
               :: "v"(ldsdst), "v"(voff), "s"(sbase)
               : "memory");
}
static __device__ __forceinline__ void wait_async() {
#if __has_builtin(__builtin_amdgcn_s_wait_asynccnt)
  __builtin_amdgcn_s_wait_asynccnt(0);
#else
  asm volatile("s_wait_asynccnt 0x0" ::: "memory");
#endif
}

// ---------------- 1. segment counts ----------------
__global__ void k_nums(const int* __restrict__ slab, float* __restrict__ nums) {
  int n = blockIdx.x * 256 + threadIdx.x;
  if (n < N_) atomicAdd(&nums[slab[n]], 1.0f);
}

// ---------------- 2. segment sum (streaming 400MB pass) ----------------
__global__ void k_segsum(const float* __restrict__ feat,
                         const int* __restrict__ slab,
                         float* __restrict__ ssum) {
  size_t i = (size_t)blockIdx.x * 256 + threadIdx.x;   // over N * D/4
  if (i >= (size_t)N_ * (D_ / 4)) return;
  int n = (int)(i / (D_ / 4));
  int c = (int)(i % (D_ / 4)) * 4;
  float4 v = *(const float4*)&feat[(size_t)n * D_ + c];
  float* dst = ssum + (size_t)slab[n] * D_ + c;
  atomicAdd(dst + 0, v.x); atomicAdd(dst + 1, v.y);
  atomicAdd(dst + 2, v.z); atomicAdd(dst + 3, v.w);
}

// ---------------- 3. mean + simm ----------------
__global__ void k_finalize(float* __restrict__ ssum,
                           const float* __restrict__ nums,
                           float* __restrict__ simm) {
  __shared__ float red[256];
  int s = blockIdx.x, t = threadIdx.x;
  float cnt = nums[s];
  float inv = cnt > 0.0f ? 1.0f / cnt : 1.0f;
  float acc = 0.0f;
#pragma unroll
  for (int j = 0; j < 8; ++j) {
    size_t o = (size_t)s * D_ + t + j * 256;
    float m = ssum[o] * inv;
    ssum[o] = m;
    acc += m * m;
  }
  red[t] = acc; __syncthreads();
  for (int st = 128; st > 0; st >>= 1) {
    if (t < st) red[t] += red[t + st];
    __syncthreads();
  }
  if (t == 0) simm[s] = red[0];
}

// ---------------- 4. column weights exp(pred[:,:,1]) ----------------
__global__ void k_colw(const float* __restrict__ pred, float* __restrict__ colw) {
  int i = blockIdx.x * 256 + threadIdx.x;
  if (i < B_ * K_) colw[i] = __expf(pred[i * 2 + 1]);
}

// ---------------- 5. gather sub_feat rows, bf16 + normalized bf16 ----------------
__global__ void k_gather(const int* __restrict__ indexes,
                         const float* __restrict__ feat,
                         const int* __restrict__ slab,
                         const int* __restrict__ knn,
                         const float* __restrict__ mean,
                         unsigned short* __restrict__ sf16,
                         unsigned short* __restrict__ xn16) {
  __shared__ float red[256];
  int bk = blockIdx.x;
  int b = bk >> 6, k = bk & 63;
  int t = threadIdx.x;
  const float* src = (k == 0) ? (feat + (size_t)indexes[b] * D_)
                              : (mean + (size_t)slab[knn[bk]] * D_);
  float r[8]; float ss = 0.0f;
#pragma unroll
  for (int j = 0; j < 8; ++j) { r[j] = src[t + j * 256]; ss += r[j] * r[j]; }
  red[t] = ss; __syncthreads();
  for (int st = 128; st > 0; st >>= 1) {
    if (t < st) red[t] += red[t + st];
    __syncthreads();
  }
  float rn = 1.0f / sqrtf(red[0]);
#pragma unroll
  for (int j = 0; j < 8; ++j) {
    size_t o = (size_t)bk * D_ + t + j * 256;
    sf16[o] = f2b(r[j]);
    xn16[o] = f2b(r[j] * rn);
  }
}

// ---------------- 6. x = xn - xn[:,0:1] (in place per batch) ----------------
__global__ void k_sub0(unsigned short* __restrict__ xn16) {
  __shared__ alignas(16) unsigned short l0[D_];
  int b = blockIdx.x, t = threadIdx.x;
  // async snapshot of row 0 (2048 bf16 = 256 lanes x 16B)
  async_ld_b128(lds_off(&l0[t * 8]),
                xn16 + (size_t)b * K_ * D_,
                (unsigned)(t * 16));
  wait_async();
  __syncthreads();
  for (int k = 0; k < K_; ++k) {
#pragma unroll
    for (int j = 0; j < 8; ++j) {
      int d = t + j * 256;
      size_t o = ((size_t)b * K_ + k) * D_ + d;
      xn16[o] = f2b(b2f(xn16[o]) - b2f(l0[d]));
    }
  }
}

// ---------------- 7. Gram: Araw[b] = X Xt  (X = sf16[b] 64x2048) ----------------
// Double-buffered async staging: DMA of tile i+1 overlaps WMMA on tile i.
__global__ void k_gram(const unsigned short* __restrict__ SF,
                       float* __restrict__ Araw) {
  __shared__ alignas(16) unsigned short lX[2][64 * 32];
  int b = blockIdx.x;
  int t = threadIdx.x;
  int wave = t >> 5, lane = t & 31;
  int half = lane >> 4, r = lane & 15;
  int id0 = wave * 2, id1 = id0 + 1;
  int tm0 = id0 >> 2, tn0 = id0 & 3;
  int tm1 = id1 >> 2, tn1 = id1 & 3;
  int row = t >> 2, seg = t & 3;
  const unsigned short* base = SF + (size_t)b * K_ * D_;
  unsigned gof0 = (unsigned)((row * D_ + seg * 8) * 2);
  unsigned ldst[2] = { lds_off(&lX[0][row * 32 + seg * 8]),
                       lds_off(&lX[1][row * 32 + seg * 8]) };
  v8f c0 = {}, c1 = {};
  const int nIter = D_ / 32;
  async_ld_b128(ldst[0], base, gof0);                 // prologue: tile 0
  for (int i = 0; i < nIter; ++i) {
    wait_async();
    __syncthreads();                                   // publish buf[i&1]
    if (i + 1 < nIter)                                 // DMA next tile, overlapped
      async_ld_b128(ldst[(i + 1) & 1], base, gof0 + (unsigned)((i + 1) * 64));
    const unsigned short* buf = lX[i & 1];
    v16bf a0 = frag_ld(&buf[(tm0 * 16 + r) * 32], half);
    v16bf b0 = frag_ld(&buf[(tn0 * 16 + r) * 32], half);
    c0 = wmma_bf16(a0, b0, c0);
    v16bf a1 = frag_ld(&buf[(tm1 * 16 + r) * 32], half);
    v16bf b1 = frag_ld(&buf[(tn1 * 16 + r) * 32], half);
    c1 = wmma_bf16(a1, b1, c1);
    __syncthreads();                                   // retire buf[i&1]
  }
#pragma unroll
  for (int i = 0; i < 8; ++i) {
    Araw[(size_t)b * 4096 + (tm0 * 16 + i + half * 8) * 64 + tn0 * 16 + r] = c0[i];
    Araw[(size_t)b * 4096 + (tm1 * 16 + i + half * 8) * 64 + tn1 * 16 + r] = c1[i];
  }
}

// ---------------- 8. A = softmax(Araw * colw) -> bf16 ----------------
__global__ void k_softmaxA(const float* __restrict__ Araw,
                           const float* __restrict__ colw,
                           unsigned short* __restrict__ A16) {
  __shared__ float sm[64];
  int bi = blockIdx.x;
  int b = bi >> 6, i = bi & 63;
  int j = threadIdx.x;                        // 64 threads
  float v = Araw[(size_t)b * 4096 + i * 64 + j] * colw[b * 64 + j];
  sm[j] = v; __syncthreads();
  for (int s = 32; s > 0; s >>= 1) { if (j < s) sm[j] = fmaxf(sm[j], sm[j + s]); __syncthreads(); }
  float m = sm[0]; __syncthreads();
  float e = __expf(v - m);
  sm[j] = e; __syncthreads();
  for (int s = 32; s > 0; s >>= 1) { if (j < s) sm[j] += sm[j + s]; __syncthreads(); }
  float inv = 1.0f / sm[0];
  A16[(size_t)b * 4096 + i * 64 + j] = f2b(e * inv);
}

// ---------------- 9. batched agg = A16[b] (64x64) @ X[b] (64xldx) -> bf16 ----------------
__global__ void k_aggA(const unsigned short* __restrict__ A16,
                       const unsigned short* __restrict__ X,
                       unsigned short* __restrict__ Out, int ldx) {
  __shared__ alignas(16) unsigned short lA[64 * 64];
  __shared__ alignas(16) unsigned short lB[64 * 64];   // [n][k] transposed
  int b = blockIdx.y;
  int colbase = blockIdx.x * 64;
  int t = threadIdx.x;
  int wave = t >> 5, lane = t & 31;
  int half = lane >> 4, r = lane & 15;
  int id0 = wave * 2, id1 = id0 + 1;
  int tm0 = id0 >> 2, tn0 = id0 & 3;
  int tm1 = id1 >> 2, tn1 = id1 & 3;
  // A (64x64, contiguous): two async b128 per thread
  async_ld_b128(lds_off(&lA[t * 8]), A16 + (size_t)b * 4096, (unsigned)(t * 16));
  async_ld_b128(lds_off(&lA[(t + 256) * 8]), A16 + (size_t)b * 4096,
                (unsigned)((t + 256) * 16));
  {
    int k = t >> 2, nseg = t & 3;              // 16 columns per thread
    const unsigned short* sp = X + ((size_t)b * K_ + k) * ldx + colbase + nseg * 16;
    float4 v0 = *(const float4*)sp;
    float4 v1 = *(const float4*)(sp + 8);
    const unsigned short* s0 = (const unsigned short*)&v0;
    const unsigned short* s1 = (const unsigned short*)&v1;
#pragma unroll
    for (int j = 0; j < 8; ++j) {
      lB[(nseg * 16 + j) * 64 + k] = s0[j];
      lB[(nseg * 16 + 8 + j) * 64 + k] = s1[j];
    }
  }
  wait_async();
  __syncthreads();
  v8f c0 = {}, c1 = {};
#pragma unroll
  for (int kk = 0; kk < 64; kk += 32) {
    v16bf a0 = frag_ld(&lA[(tm0 * 16 + r) * 64 + kk], half);
    v16bf b0 = frag_ld(&lB[(tn0 * 16 + r) * 64 + kk], half);
    c0 = wmma_bf16(a0, b0, c0);
    v16bf a1 = frag_ld(&lA[(tm1 * 16 + r) * 64 + kk], half);
    v16bf b1 = frag_ld(&lB[(tn1 * 16 + r) * 64 + kk], half);
    c1 = wmma_bf16(a1, b1, c1);
  }
#pragma unroll
  for (int i = 0; i < 8; ++i) {
    Out[((size_t)b * K_ + tm0 * 16 + i + half * 8) * ldx + colbase + tn0 * 16 + r] = f2b(c0[i]);
    Out[((size_t)b * K_ + tm1 * 16 + i + half * 8) * ldx + colbase + tn1 * 16 + r] = f2b(c1[i]);
  }
}

// ---------------- 10. generic bf16 GEMM: C(f32) (+)= A16[MxKd] @ B16[KdxNd] ----------------
// Block tile 128x64, 8 waves, 4 WMMA tiles per wave (tm = wave, tn = 0..3).
// Double-buffered LDS; A tiles staged via async DMA, B via coalesced
// load + transposed scatter, both overlapped with WMMA on previous tile.
__global__ void k_gemm(const unsigned short* __restrict__ A16,
                       const unsigned short* __restrict__ B16,
                       float* __restrict__ C,
                       int Kd, int ldb, int beta) {
  __shared__ alignas(16) unsigned short lA[2][128 * 32];
  __shared__ alignas(16) unsigned short lB[2][64 * 32];   // [n][k]
  int t = threadIdx.x;
  int wave = t >> 5, lane = t & 31;
  int half = lane >> 4, r = lane & 15;
  int rowbase = blockIdx.y * 128;
  int colbase = blockIdx.x * 64;
  // A staging: 128x32 tile = 512 b128 chunks; thread t stages chunks 2t, 2t+1
  int arow = t >> 1, asg = (t & 1) * 2;
  const unsigned short* abase = A16 + (size_t)rowbase * Kd;
  int bk_ = t >> 3, bn_ = t & 7;             // B staging coords
  v8f acc[4] = {{}, {}, {}, {}};
  const int nIter = Kd / 32;

  auto stage = [&](int sel, int kk) {
    async_ld_b128(lds_off(&lA[sel][arow * 32 + asg * 8]), abase,
                  (unsigned)((arow * (size_t)Kd + kk + asg * 8) * 2));
    async_ld_b128(lds_off(&lA[sel][arow * 32 + (asg + 1) * 8]), abase,
                  (unsigned)((arow * (size_t)Kd + kk + (asg + 1) * 8) * 2));
    float4 v = *(const float4*)&B16[(size_t)(kk + bk_) * ldb + colbase + bn_ * 8];
    const unsigned short* s = (const unsigned short*)&v;
#pragma unroll
    for (int j = 0; j < 8; ++j) lB[sel][(bn_ * 8 + j) * 32 + bk_] = s[j];
  };

  stage(0, 0);                                // prologue
  for (int i = 0; i < nIter; ++i) {
    wait_async();
    __syncthreads();                          // publish buf[i&1]
    if (i + 1 < nIter) stage((i + 1) & 1, (i + 1) * 32);   // overlapped staging
    const unsigned short* bufA = lA[i & 1];
    const unsigned short* bufB = lB[i & 1];
    v16bf a = frag_ld(&bufA[(wave * 16 + r) * 32], half);
#pragma unroll
    for (int j = 0; j < 4; ++j) {
      v16bf bb = frag_ld(&bufB[(j * 16 + r) * 32], half);
      acc[j] = wmma_bf16(a, bb, acc[j]);
    }
    __syncthreads();                          // retire buf[i&1]
  }
#pragma unroll
  for (int j = 0; j < 4; ++j) {
#pragma unroll
    for (int i = 0; i < 8; ++i) {
      size_t o = (size_t)(rowbase + wave * 16 + i + half * 8) * ldb +
                 colbase + j * 16 + r;
      if (beta) C[o] += acc[j][i];
      else      C[o]  = acc[j][i];
    }
  }
}

// ---------------- 11. epilogue: bias + relu / prelu -> bf16 ----------------
__global__ void k_bias_act(const float* __restrict__ acc,
                           const float* __restrict__ bias,
                           const float* __restrict__ pa,
                           int nd, int mode,
                           unsigned short* __restrict__ out16, size_t total) {
  size_t i = (size_t)blockIdx.x * 256 + threadIdx.x;
  if (i >= total) return;
  int col = (int)(i % nd);
  float h = acc[i] + bias[col];
  h = (mode == 0) ? fmaxf(h, 0.0f) : (h >= 0.0f ? h : pa[col] * h);
  out16[i] = f2b(h);
}

// ---------------- 12. f32 -> bf16 weight conversion ----------------
__global__ void k_tobf16(const float* __restrict__ in,
                         unsigned short* __restrict__ out, size_t n) {
  size_t i = (size_t)blockIdx.x * 256 + threadIdx.x;
  if (i < n) out[i] = f2b(in[i]);
}

// ---------------- 13. logits + 2-class softmax (one wave32 per row) ----------------
__global__ void k_logits(const unsigned short* __restrict__ hc,
                         const float* __restrict__ Wc2,
                         const float* __restrict__ bc2,
                         float* __restrict__ probs) {
  int wave = threadIdx.x >> 5, lane = threadIdx.x & 31;
  int row = blockIdx.x * 8 + wave;
  float s0 = 0.0f, s1 = 0.0f;
#pragma unroll
  for (int j = 0; j < 8; ++j) {
    int k = lane + j * 32;
    float a = b2f(hc[(size_t)row * DOUT_ + k]);
    s0 += a * Wc2[k * 2 + 0];
    s1 += a * Wc2[k * 2 + 1];
  }
#pragma unroll
  for (int off = 16; off > 0; off >>= 1) {
    s0 += __shfl_xor(s0, off, 32);
    s1 += __shfl_xor(s1, off, 32);
  }
  if (lane == 0) {
    float l0 = s0 + bc2[0], l1 = s1 + bc2[1];
    float m = fmaxf(l0, l1);
    float e0 = __expf(l0 - m), e1 = __expf(l1 - m);
    float inv = 1.0f / (e0 + e1);
    probs[(size_t)row * 2 + 0] = e0 * inv;
    probs[(size_t)row * 2 + 1] = e1 * inv;
  }
}

// ---------------- host launcher ----------------
extern "C" void kernel_launch(void* const* d_in, const int* in_sizes, int n_in,
                              void* d_out, int out_size, void* d_ws, size_t ws_size,
                              hipStream_t stream) {
  (void)in_sizes; (void)n_in; (void)out_size; (void)ws_size;
  const int*   indexes = (const int*)d_in[0];
  const float* features= (const float*)d_in[1];
  const int*   sublab  = (const int*)d_in[3];
  const int*   knn     = (const int*)d_in[6];
  const float* allpred = (const float*)d_in[7];
  const float* W1      = (const float*)d_in[9];
  const float* b1      = (const float*)d_in[10];
  const float* W2      = (const float*)d_in[11];
  const float* b2      = (const float*)d_in[12];
  const float* Wc1     = (const float*)d_in[13];
  const float* bc1     = (const float*)d_in[14];
  const float* pa      = (const float*)d_in[15];
  const float* Wc2     = (const float*)d_in[16];
  const float* bc2     = (const float*)d_in[17];

  // d_out: probs[B*K*2] | simm[S] | sub_sum[S*D] | nums[S]
  float* out    = (float*)d_out;
  float* probs  = out;
  float* simm   = out + (size_t)B_ * K_ * 2;
  float* subsum = simm + S_;
  float* nums   = subsum + (size_t)S_ * D_;

  // workspace slabs (liveness-aliased)
  char* w = (char*)d_ws;
  const size_t SF = (size_t)B_ * K_ * D_ * 2;                // 64 MB
  unsigned short* sf16  = (unsigned short*)w;                 // slab0
  unsigned short* agg0  = sf16;                               // alias (after gram)
  unsigned short* xn16  = (unsigned short*)(w + SF);          // slab1
  float*          h2acc = (float*)(w + SF);                   // alias (after xn dead)
  unsigned short* h2_16 = (unsigned short*)(w + SF + 16ull*1024*1024);
  float*          hcacc = (float*)(w + SF + 24ull*1024*1024);
  unsigned short* hc16  = (unsigned short*)(w + SF + 40ull*1024*1024);
  float*          h1acc = (float*)(w + 2*SF);                 // slab2
  unsigned short* h1_16 = (unsigned short*)(w + 2*SF + 32ull*1024*1024);
  unsigned short* agg1  = (unsigned short*)(w + 2*SF + 48ull*1024*1024);
  char* w4 = w + 2*SF + 64ull*1024*1024;                      // slab4 (small stuff)
  float*          Araw  = (float*)w4;                          // 4 MB
  unsigned short* A16   = (unsigned short*)(w4 + 4194304);     // 2 MB
  float*          colw  = (float*)(w4 + 4194304 + 2097152);    // 64 KB
  unsigned short* W1_16 = (unsigned short*)(w4 + 4194304 + 2097152 + 65536);
  unsigned short* W2_16 = W1_16 + (size_t)4096 * 512;
  unsigned short* Wc1_16= W2_16 + (size_t)1024 * 256;

  // zero accumulation targets in d_out (sub_sum + nums are contiguous)
  hipMemsetAsync(subsum, 0, ((size_t)S_ * D_ + S_) * sizeof(float), stream);

  // weights -> bf16
  k_tobf16<<<8192, 256, 0, stream>>>(W1, W1_16, (size_t)4096 * 512);
  k_tobf16<<<1024, 256, 0, stream>>>(W2, W2_16, (size_t)1024 * 256);
  k_tobf16<<<256, 256, 0, stream>>>(Wc1, Wc1_16, (size_t)256 * 256);

  // segment mean + simm
  k_nums<<<(N_ + 255) / 256, 256, 0, stream>>>(sublab, nums);
  k_segsum<<<(int)(((size_t)N_ * (D_ / 4) + 255) / 256), 256, 0, stream>>>(features, sublab, subsum);
  k_finalize<<<S_, 256, 0, stream>>>(subsum, nums, simm);

  // adjacency inputs
  k_colw<<<(B_ * K_ + 255) / 256, 256, 0, stream>>>(allpred, colw);
  k_gather<<<B_ * K_, 256, 0, stream>>>(indexes, features, sublab, knn, subsum, sf16, xn16);
  k_sub0<<<B_, 256, 0, stream>>>(xn16);

  // A = softmax(X Xt * colw)
  k_gram<<<B_, 256, 0, stream>>>(sf16, Araw);
  k_softmaxA<<<B_ * K_, 64, 0, stream>>>(Araw, colw, A16);

  // GCN layer 1: h1 = relu(cat(x, A@x) @ W1 + b1)
  k_aggA<<<dim3(D_ / 64, B_), 256, 0, stream>>>(A16, xn16, agg0, D_);
  k_gemm<<<dim3(NHID_ / 64, (B_ * K_) / 128), 256, 0, stream>>>(xn16, W1_16, h1acc, D_, NHID_, 0);
  k_gemm<<<dim3(NHID_ / 64, (B_ * K_) / 128), 256, 0, stream>>>(agg0, W1_16 + (size_t)D_ * NHID_, h1acc, D_, NHID_, 1);
  k_bias_act<<<(int)(((size_t)B_ * K_ * NHID_ + 255) / 256), 256, 0, stream>>>(
      h1acc, b1, b1, NHID_, 0, h1_16, (size_t)B_ * K_ * NHID_);

  // GCN layer 2: h2 = relu(cat(h1, A@h1) @ W2 + b2)
  k_aggA<<<dim3(NHID_ / 64, B_), 256, 0, stream>>>(A16, h1_16, agg1, NHID_);
  k_gemm<<<dim3(DOUT_ / 64, (B_ * K_) / 128), 256, 0, stream>>>(h1_16, W2_16, h2acc, NHID_, DOUT_, 0);
  k_gemm<<<dim3(DOUT_ / 64, (B_ * K_) / 128), 256, 0, stream>>>(agg1, W2_16 + (size_t)NHID_ * DOUT_, h2acc, NHID_, DOUT_, 1);
  k_bias_act<<<(int)(((size_t)B_ * K_ * DOUT_ + 255) / 256), 256, 0, stream>>>(
      h2acc, b2, b2, DOUT_, 0, h2_16, (size_t)B_ * K_ * DOUT_);

  // classifier: prelu(h2 @ Wc1 + bc1) @ Wc2 + bc2 -> softmax
  k_gemm<<<dim3(DOUT_ / 64, (B_ * K_) / 128), 256, 0, stream>>>(h2_16, Wc1_16, hcacc, DOUT_, DOUT_, 0);
  k_bias_act<<<(int)(((size_t)B_ * K_ * DOUT_ + 255) / 256), 256, 0, stream>>>(
      hcacc, bc1, pa, DOUT_, 1, hc16, (size_t)B_ * K_ * DOUT_);
  k_logits<<<(B_ * K_) / 8, 256, 0, stream>>>(hc16, Wc2, bc2, probs);
}
